// EnhancedBiologicalSplatAttentionLayer_49744311222296
// MI455X (gfx1250) — matrix operand order repaired and checked by reading, not compiled, not run on hardware
//
#include <hip/hip_runtime.h>
#include <hip/hip_bf16.h>

// ---------- types ----------
typedef __attribute__((ext_vector_type(16))) __bf16 v16bf;
typedef __attribute__((ext_vector_type(8)))  float  v8f;
typedef __attribute__((ext_vector_type(4)))  unsigned int uint4v;

struct FragBF { union { uint4v q[2]; v16bf v; }; };

#define BB 4
#define SS 2048
#define DD 1024
#define KK 64
#define BS 8192   // B*S

#define LDS_ROW 40   // padded row stride in halves (80B): b128-aligned, conflict-free frag reads

__device__ __forceinline__ unsigned short f2bf(float f) {
    unsigned int u = __float_as_uint(f);
    unsigned int r = (u + 0x7FFFu + ((u >> 16) & 1u)) >> 16;
    return (unsigned short)r;
}

// ---------- fragment loads (CDNA5 wave32 WMMA layouts) ----------
// A 16x32 bf16: lanes 0-15 row m, K = {0..7, 16..23}; lanes 16-31 row m, K = {8..15, 24..31}
__device__ __forceinline__ v16bf load_a_frag(const unsigned short* A, int lda,
                                             int m0, int k0, int lane) {
    int row = m0 + (lane & 15);
    int hi  = lane >> 4;
    const unsigned short* p = A + (size_t)row * lda + k0 + hi * 8;
    FragBF f;
    f.q[0] = *reinterpret_cast<const uint4v*>(p);       // K +0..7
    f.q[1] = *reinterpret_cast<const uint4v*>(p + 16);  // K +16..23
    return f.v;
}
// B 32x16 bf16 (from row-major W[N][K]): lanes 0-15 col n, K = {0..7, 16..23}; lanes 16-31 col n, K = {8..15, 24..31}
__device__ __forceinline__ v16bf load_b_frag(const unsigned short* W, int ldw,
                                             int n0, int k0, int lane) {
    int row = n0 + (lane & 15);
    int hi  = lane >> 4;
    const unsigned short* p = W + (size_t)row * ldw + k0 + hi * 8;
    FragBF f;
    f.q[0] = *reinterpret_cast<const uint4v*>(p);       // K +0..7
    f.q[1] = *reinterpret_cast<const uint4v*>(p + 16);  // K +16..23
    return f.v;
}
// same layouts, reading a 32-halves-wide K-slab staged in LDS (row stride LDS_ROW halves)
__device__ __forceinline__ v16bf lds_a_frag(const unsigned short* tile, int m0, int lane) {
    int row = m0 + (lane & 15);
    int hi  = lane >> 4;
    const unsigned short* p = tile + row * LDS_ROW + hi * 8;
    FragBF f;
    f.q[0] = *reinterpret_cast<const uint4v*>(p);
    f.q[1] = *reinterpret_cast<const uint4v*>(p + 16);
    return f.v;
}
__device__ __forceinline__ v16bf lds_b_frag(const unsigned short* tile, int n0, int lane) {
    int row = n0 + (lane & 15);
    int hi  = lane >> 4;
    const unsigned short* p = tile + row * LDS_ROW + hi * 8;
    FragBF f;
    f.q[0] = *reinterpret_cast<const uint4v*>(p);
    f.q[1] = *reinterpret_cast<const uint4v*>(p + 16);
    return f.v;
}

// ---------- shared epilogue: store one 64x64 wave tile ----------
// mode 0: f32 normal; mode 1: bf16 transposed (C^T, ldc = ldT, contiguous b128/lane); mode 2: bf16 normal
__device__ __forceinline__ void store_tile(const v8f acc[4][4], void* __restrict__ Cout,
                                           int b, long long strideC, int ldc,
                                           int m0, int n0, int lane, int mode) {
    const int nl = lane & 15, hi = lane >> 4;
    if (mode == 0) {
        float* C = (float*)Cout + (size_t)b * strideC;
#pragma unroll
        for (int i = 0; i < 4; ++i)
#pragma unroll
            for (int j = 0; j < 4; ++j) {
                int mbase = m0 + i * 16 + hi * 8;
                int col   = n0 + j * 16 + nl;
#pragma unroll
                for (int r = 0; r < 8; ++r)
                    C[(size_t)(mbase + r) * ldc + col] = acc[i][j][r];
            }
    } else if (mode == 1) {
        unsigned short* C = (unsigned short*)Cout + (size_t)b * strideC;
#pragma unroll
        for (int i = 0; i < 4; ++i)
#pragma unroll
            for (int j = 0; j < 4; ++j) {
                size_t base = (size_t)(n0 + j * 16 + nl) * ldc + (m0 + i * 16 + hi * 8);
                unsigned int d0 = (unsigned)f2bf(acc[i][j][0]) | ((unsigned)f2bf(acc[i][j][1]) << 16);
                unsigned int d1 = (unsigned)f2bf(acc[i][j][2]) | ((unsigned)f2bf(acc[i][j][3]) << 16);
                unsigned int d2 = (unsigned)f2bf(acc[i][j][4]) | ((unsigned)f2bf(acc[i][j][5]) << 16);
                unsigned int d3 = (unsigned)f2bf(acc[i][j][6]) | ((unsigned)f2bf(acc[i][j][7]) << 16);
                uint4v qv = {d0, d1, d2, d3};
                *reinterpret_cast<uint4v*>(C + base) = qv;
            }
    } else {
        unsigned short* C = (unsigned short*)Cout + (size_t)b * strideC;
#pragma unroll
        for (int i = 0; i < 4; ++i)
#pragma unroll
            for (int j = 0; j < 4; ++j) {
                int mbase = m0 + i * 16 + hi * 8;
                int col   = n0 + j * 16 + nl;
#pragma unroll
                for (int r = 0; r < 8; ++r)
                    C[(size_t)(mbase + r) * ldc + col] = f2bf(acc[i][j][r]);
            }
    }
}

// ---------- single-wave GEMM (small K / small M stages): C = A(MxKin) * W(NxKin)^T ----------
__global__ __launch_bounds__(32)
void gemm_wmma_kernel(const unsigned short* __restrict__ A, long long strideA, int lda,
                      const unsigned short* __restrict__ W, long long strideW, int ldw,
                      void* __restrict__ Cout, long long strideC, int ldc,
                      int Kin, int mode) {
    const int lane = threadIdx.x & 31;
    const int m0 = blockIdx.x * 64;
    const int n0 = blockIdx.y * 64;
    const int b  = blockIdx.z;
    A += (size_t)b * strideA;
    W += (size_t)b * strideW;

    v8f acc[4][4];
#pragma unroll
    for (int i = 0; i < 4; ++i)
#pragma unroll
        for (int j = 0; j < 4; ++j)
            acc[i][j] = (v8f){0.f,0.f,0.f,0.f,0.f,0.f,0.f,0.f};

    for (int k0 = 0; k0 < Kin; k0 += 32) {
        v16bf af[4], bfg[4];
#pragma unroll
        for (int i = 0; i < 4; ++i)
            af[i] = load_a_frag(A, lda, m0 + i * 16, k0, lane);
#pragma unroll
        for (int j = 0; j < 4; ++j)
            bfg[j] = load_b_frag(W, ldw, n0 + j * 16, k0, lane);
        if (k0 + 32 < Kin) {  // global_prefetch_b8 next K-tile
            __builtin_prefetch(A + (size_t)(m0 + (lane & 15)) * lda + k0 + 32, 0, 1);
            __builtin_prefetch(W + (size_t)(n0 + (lane & 15)) * ldw + k0 + 32, 0, 1);
        }
#pragma unroll
        for (int i = 0; i < 4; ++i)
#pragma unroll
            for (int j = 0; j < 4; ++j)
                acc[i][j] = __builtin_amdgcn_wmma_f32_16x16x32_bf16(
                    false, af[i], false, bfg[j], (short)0, acc[i][j], false, false);
    }
    store_tile(acc, Cout, b, strideC, ldc, m0, n0, lane, mode);
}

// ---------- 4-wave, LDS double-buffered GEMM for the big K=1024 stages ----------
// WG tile 128x128 (waves 2x2 of 64x64); per K-step stage A(128x32)+B(128x32) bf16 in LDS.
// Requires M%128==0, N%128==0, Kin%32==0.
__global__ __launch_bounds__(128)
void gemm_wmma_lds_kernel(const unsigned short* __restrict__ A, long long strideA, int lda,
                          const unsigned short* __restrict__ W, long long strideW, int ldw,
                          void* __restrict__ Cout, long long strideC, int ldc,
                          int Kin, int mode) {
    __shared__ __align__(16) unsigned short Abuf[2][128 * LDS_ROW];
    __shared__ __align__(16) unsigned short Bbuf[2][128 * LDS_ROW];

    const int t    = threadIdx.x;          // 0..127
    const int lane = t & 31;
    const int wave = t >> 5;
    const int wm   = wave >> 1, wn = wave & 1;
    const int m0   = blockIdx.x * 128;
    const int n0   = blockIdx.y * 128;
    const int b    = blockIdx.z;

    const unsigned short* Ar = A + (size_t)b * strideA + (size_t)(m0 + t) * lda; // my A row
    const unsigned short* Wr = W + (size_t)b * strideW + (size_t)(n0 + t) * ldw; // my B row

    v8f acc[4][4];
#pragma unroll
    for (int i = 0; i < 4; ++i)
#pragma unroll
        for (int j = 0; j < 4; ++j)
            acc[i][j] = (v8f){0.f,0.f,0.f,0.f,0.f,0.f,0.f,0.f};

    uint4v ra[4], rb[4];
    // prologue: stage K-step 0
    {
        const uint4v* pa = reinterpret_cast<const uint4v*>(Ar);
        const uint4v* pw = reinterpret_cast<const uint4v*>(Wr);
#pragma unroll
        for (int c = 0; c < 4; ++c) { ra[c] = pa[c]; rb[c] = pw[c]; }
        uint4v* da = reinterpret_cast<uint4v*>(&Abuf[0][t * LDS_ROW]);
        uint4v* db = reinterpret_cast<uint4v*>(&Bbuf[0][t * LDS_ROW]);
#pragma unroll
        for (int c = 0; c < 4; ++c) { da[c] = ra[c]; db[c] = rb[c]; }
    }
    __syncthreads();

    const int nsteps = Kin >> 5;
    for (int kt = 0; kt < nsteps; ++kt) {
        const unsigned short* At = Abuf[kt & 1];
        const unsigned short* Bt = Bbuf[kt & 1];
        if (kt + 1 < nsteps) {  // issue next-slab global loads; overlap with WMMA below
            const uint4v* pa = reinterpret_cast<const uint4v*>(Ar + (kt + 1) * 32);
            const uint4v* pw = reinterpret_cast<const uint4v*>(Wr + (kt + 1) * 32);
#pragma unroll
            for (int c = 0; c < 4; ++c) { ra[c] = pa[c]; rb[c] = pw[c]; }
        }
        v16bf af[4], bfg[4];
#pragma unroll
        for (int i = 0; i < 4; ++i) af[i]  = lds_a_frag(At, wm * 64 + i * 16, lane);
#pragma unroll
        for (int j = 0; j < 4; ++j) bfg[j] = lds_b_frag(Bt, wn * 64 + j * 16, lane);
#pragma unroll
        for (int i = 0; i < 4; ++i)
#pragma unroll
            for (int j = 0; j < 4; ++j)
                acc[i][j] = __builtin_amdgcn_wmma_f32_16x16x32_bf16(
                    false, af[i], false, bfg[j], (short)0, acc[i][j], false, false);
        if (kt + 1 < nsteps) {  // write next buffer (disjoint from 'cur'), then one barrier
            uint4v* da = reinterpret_cast<uint4v*>(&Abuf[(kt + 1) & 1][t * LDS_ROW]);
            uint4v* db = reinterpret_cast<uint4v*>(&Bbuf[(kt + 1) & 1][t * LDS_ROW]);
#pragma unroll
            for (int c = 0; c < 4; ++c) { da[c] = ra[c]; db[c] = rb[c]; }
            __syncthreads();
        }
    }
    store_tile(acc, Cout, b, strideC, ldc, m0 + wm * 64, n0 + wn * 64, lane, mode);
}

// ---------- helpers ----------
__global__ void f32_to_bf16_kernel(const float* __restrict__ in,
                                   unsigned short* __restrict__ out, int n) {
    int i = blockIdx.x * blockDim.x + threadIdx.x;
    if (i < n) out[i] = f2bf(in[i]);
}

// one wave per token row: xsq[t] = sum_d x[t][d]^2
__global__ __launch_bounds__(256)
void xsq_kernel(const float* __restrict__ x, float* __restrict__ xsq) {
    int wave = threadIdx.x >> 5, lane = threadIdx.x & 31;
    int row = blockIdx.x * 8 + wave;
    const float* xr = x + (size_t)row * DD;
    float s = 0.f;
    for (int i = lane; i < DD; i += 32) { float t = xr[i]; s += t * t; }
#pragma unroll
    for (int off = 16; off; off >>= 1) s += __shfl_xor(s, off);
    if (lane == 0) xsq[row] = s;
}

// csq[k] = |c_k|^2 ; inv2s2[k] = 0.5 / clip(exp(ls),0.1,2)^2
__global__ __launch_bounds__(64)
void csq_kernel(const float* __restrict__ centers, const float* __restrict__ log_scales,
                float* __restrict__ csq, float* __restrict__ inv2s2) {
    int k = threadIdx.x;
    const float* c = centers + (size_t)k * DD;
    float s = 0.f;
    for (int i = 0; i < DD; ++i) { float t = c[i]; s += t * t; }
    csq[k] = s;
    float sc = expf(log_scales[k]);
    sc = fminf(fmaxf(sc, 0.1f), 2.0f);
    inv2s2[k] = 0.5f / (sc * sc);
}

// one wave per token: aff = exp(-dist*inv2s2), normalize over K=64, emit aff (bf16) + affT (bf16)
__global__ __launch_bounds__(256)
void aff_kernel(const float* __restrict__ xc, const float* __restrict__ xsq,
                const float* __restrict__ csq, const float* __restrict__ inv2s2,
                unsigned short* __restrict__ aff_bf, unsigned short* __restrict__ affT_bf) {
    int wave = threadIdx.x >> 5, lane = threadIdx.x & 31;
    int token = blockIdx.x * 8 + wave;
    const float* xcr = xc + (size_t)token * KK;
    float xs = xsq[token];
    float d0 = xs - 2.f * xcr[lane]      + csq[lane];
    float d1 = xs - 2.f * xcr[lane + 32] + csq[lane + 32];
    float a0 = expf(-d0 * inv2s2[lane]);
    float a1 = expf(-d1 * inv2s2[lane + 32]);
    float s = a0 + a1;
#pragma unroll
    for (int off = 16; off; off >>= 1) s += __shfl_xor(s, off);
    float inv = 1.f / (s + 1e-8f);
    a0 *= inv; a1 *= inv;
    aff_bf[(size_t)token * KK + lane]      = f2bf(a0);
    aff_bf[(size_t)token * KK + lane + 32] = f2bf(a1);
    int b = token >> 11, si = token & 2047;
    affT_bf[((size_t)b * KK + lane)      * SS + si] = f2bf(a0);
    affT_bf[((size_t)b * KK + lane + 32) * SS + si] = f2bf(a1);
}

// ---------- host ----------
extern "C" void kernel_launch(void* const* d_in, const int* in_sizes, int n_in,
                              void* d_out, int out_size, void* d_ws, size_t ws_size,
                              hipStream_t stream) {
    const float* x   = (const float*)d_in[0];  // [B,S,D]
    const float* ctr = (const float*)d_in[1];  // [K,D]
    const float* ls  = (const float*)d_in[2];  // [K]
    const float* wv  = (const float*)d_in[3];  // [D,D]
    const float* wo  = (const float*)d_in[4];  // [D,D]
    float* out = (float*)d_out;                // [B,S,D] f32

    char* p = (char*)d_ws;
    auto carve = [&](size_t bytes) { void* r = (void*)p; p += (bytes + 255) & ~(size_t)255; return r; };
    unsigned short* xb      = (unsigned short*)carve((size_t)BS * DD * 2);   // x bf16
    unsigned short* wvb     = (unsigned short*)carve((size_t)DD * DD * 2);
    unsigned short* wob     = (unsigned short*)carve((size_t)DD * DD * 2);
    unsigned short* cb      = (unsigned short*)carve((size_t)KK * DD * 2);
    float*          xsq     = (float*)carve((size_t)BS * 4);
    float*          csq     = (float*)carve((size_t)KK * 4);
    float*          inv2s2  = (float*)carve((size_t)KK * 4);
    float*          xc      = (float*)carve((size_t)BS * KK * 4);
    unsigned short* aff_bf  = (unsigned short*)carve((size_t)BS * KK * 2);
    unsigned short* affT_bf = (unsigned short*)carve((size_t)BB * KK * SS * 2);
    unsigned short* vT      = (unsigned short*)carve((size_t)BB * DD * SS * 2);  // [b][d][s]
    unsigned short* stT     = (unsigned short*)carve((size_t)BB * DD * KK * 2);  // [b][d][k]
    unsigned short* tok     = (unsigned short*)carve((size_t)BS * DD * 2);       // [b*s][d]

    // 1) f32 -> bf16 conversions
    f32_to_bf16_kernel<<<(BS * DD + 255) / 256, 256, 0, stream>>>(x,   xb,  BS * DD);
    f32_to_bf16_kernel<<<(DD * DD + 255) / 256, 256, 0, stream>>>(wv,  wvb, DD * DD);
    f32_to_bf16_kernel<<<(DD * DD + 255) / 256, 256, 0, stream>>>(wo,  wob, DD * DD);
    f32_to_bf16_kernel<<<(KK * DD + 255) / 256, 256, 0, stream>>>(ctr, cb,  KK * DD);

    // 2) row norms + center norms/scales
    xsq_kernel<<<BS / 8, 256, 0, stream>>>(x, xsq);
    csq_kernel<<<1, 64, 0, stream>>>(ctr, ls, csq, inv2s2);

    // 3) xc = x @ centers^T   [8192 x 64], f32 normal (N=64 -> single-wave kernel)
    gemm_wmma_kernel<<<dim3(BS / 64, KK / 64, 1), 32, 0, stream>>>(
        xb, 0, DD, cb, 0, DD, xc, 0, KK, DD, 0);

    // 4) affinities (f32 math, matches reference incl. underflow), emit aff + affT bf16
    aff_kernel<<<BS / 8, 256, 0, stream>>>(xc, xsq, csq, inv2s2, aff_bf, affT_bf);

    // 5) v = x @ Wv^T (big GEMM, LDS-staged), stored transposed per batch: vT[b][d][s]
    gemm_wmma_lds_kernel<<<dim3(SS / 128, DD / 128, BB), 128, 0, stream>>>(
        xb, (long long)SS * DD, DD, wvb, 0, DD, vT, (long long)DD * SS, SS, DD, 1);

    // 6) states[b][k][d] = affT[b] (64xS) @ vT[b] (DxS)^T, stored transposed: stT[b][d][k]
    //    (M=64 -> single-wave kernel)
    gemm_wmma_kernel<<<dim3(KK / 64, DD / 64, BB), 32, 0, stream>>>(
        affT_bf, (long long)KK * SS, SS, vT, (long long)DD * SS, SS,
        stT, (long long)DD * KK, KK, SS, 1);

    // 7) tok[b][s][d] = aff[b] (Sx64) @ stT[b] (Dx64)^T, bf16 normal (Kin=64 -> single-wave)
    gemm_wmma_kernel<<<dim3(SS / 64, DD / 64, BB), 32, 0, stream>>>(
        aff_bf, (long long)SS * KK, KK, stT, (long long)DD * KK, KK,
        tok, (long long)SS * DD, DD, KK, 2);

    // 8) out = tok @ Wo^T (big GEMM, LDS-staged), f32 -> d_out
    gemm_wmma_lds_kernel<<<dim3(BS / 128, DD / 128, 1), 128, 0, stream>>>(
        tok, 0, DD, wob, 0, DD, out, 0, DD, DD, 0);
}
